// RetICLBase_83339545411715
// MI455X (gfx1250) — compile-verified, matrix operand order checked by reading, not run on hardware
//
#include <hip/hip_runtime.h>
#include <hip/hip_bf16.h>

// Problem dims
#define BDIM 64
#define TDIM 4
#define EDIM 768
#define HDIM 768
#define NDIM 100000
#define CDIM 2
#define MTOT 768            // (C+1) * B*T = 3 * 256 rows, head-major == output layout

// GEMM tiling
#define BM 128
#define BN 128
#define BK 32
#define LDA 40              // BK + 8 pad (bf16 elems) -> 80B rows, 16B aligned

typedef __attribute__((ext_vector_type(16))) __bf16 v16bf;
typedef __attribute__((ext_vector_type(8)))  float  v8f;

union FragBF { uint4 u[2]; v16bf v; };
union AccF   { v8f v; float f[8]; };

static __device__ inline __bf16 f2bf(float f) {
    union { float f; unsigned u; } s; s.f = f;
    unsigned r = s.u + 0x7FFFu + ((s.u >> 16) & 1u);   // round to nearest even
    union { unsigned short u; __bf16 b; } d; d.u = (unsigned short)(r >> 16);
    return d.b;
}

// Pack two f32 -> two bf16 in one dword: round-to-nearest bias adds + v_perm_b32
// result: low16 = bf16(a), high16 = bf16(b)
static __device__ inline unsigned pack2bf(float a, float b) {
    union { float f; unsigned u; } ua, ub; ua.f = a; ub.f = b;
    const unsigned ra = ua.u + 0x8000u;
    const unsigned rb = ub.u + 0x8000u;
    // {S0,S1} byte pool: S0 = bytes 7:4, S1 = bytes 3:0; sel 0x07060302 ->
    // D = { S0[31:16], S1[31:16] }
    return __builtin_amdgcn_perm(rb, ra, 0x07060302u);
}

// ---------------------------------------------------------------------------
// Kernel 1: lat[b,t,h] = tanh(cumsum_t( [cur; ex] @ w_in )),  t = 0..3
// (ex[b,3] is never needed since the reference drops the last step)
// ---------------------------------------------------------------------------
__global__ __launch_bounds__(256) void latent_kernel(
    const float* __restrict__ cur, const float* __restrict__ ex,
    const float* __restrict__ w_in, float* __restrict__ lat)
{
    const int h = blockIdx.x * 256 + threadIdx.x;   // 0..767 (grid.x = 3)
    const int b = blockIdx.y;
    const float* c  = cur + (size_t)b * EDIM;
    const float* e0 = ex  + (size_t)b * TDIM * EDIM;
    float ac = 0.f, a0 = 0.f, a1 = 0.f, a2 = 0.f;
    for (int e = 0; e < EDIM; ++e) {
        const float w = w_in[(size_t)e * HDIM + h];  // coalesced across threads
        ac += c[e] * w;
        a0 += e0[e] * w;
        a1 += e0[EDIM + e] * w;
        a2 += e0[2 * EDIM + e] * w;
    }
    float* L = lat + (size_t)b * TDIM * HDIM + h;
    float s = ac;  L[0]        = tanhf(s);
    s += a0;       L[HDIM]     = tanhf(s);
    s += a1;       L[2 * HDIM] = tanhf(s);
    s += a2;       L[3 * HDIM] = tanhf(s);
}

// ---------------------------------------------------------------------------
// Kernel 2: Abig[row, e] (bf16), row = head*256 + (b*T+t)
//   head 0: lat @ bilinear ; head 1,2: lat @ critic_bilinear[head-1]
// ---------------------------------------------------------------------------
__global__ __launch_bounds__(256) void aproj_kernel(
    const float* __restrict__ lat, const float* __restrict__ bilinear,
    const float* __restrict__ critic_bilinear, __bf16* __restrict__ Abig)
{
    __shared__ float lrow[HDIM];
    const int row  = blockIdx.x;        // 0..767
    const int head = row >> 8;
    const int bt   = row & 255;
    const float* W = (head == 0) ? bilinear
                                 : (critic_bilinear + (size_t)(head - 1) * HDIM * EDIM);
    const int tid = threadIdx.x;
    for (int i = tid; i < HDIM; i += 256) lrow[i] = lat[(size_t)bt * HDIM + i];
    __syncthreads();
    float s0 = 0.f, s1 = 0.f, s2 = 0.f;
    for (int hh = 0; hh < HDIM; ++hh) {
        const float lv = lrow[hh];
        const float* wr = W + (size_t)hh * EDIM;
        s0 += lv * wr[tid];
        s1 += lv * wr[tid + 256];
        s2 += lv * wr[tid + 512];
    }
    __bf16* dst = Abig + (size_t)row * EDIM;
    dst[tid]       = f2bf(s0);
    dst[tid + 256] = f2bf(s1);
    dst[tid + 512] = f2bf(s2);
}

// ---------------------------------------------------------------------------
// Kernel 3: out[row, n] = Abig[row, :] . bank[n, :] + bias(row)
//   768 x 100000 GEMM over K=768 on v_wmma_f32_16x16x32_bf16.
//   Bank tiles converted f32->bf16 (v_perm packing), staged through LDS.
// ---------------------------------------------------------------------------
__global__ __launch_bounds__(256) void gemm_bank_kernel(
    const __bf16* __restrict__ Abig, const float* __restrict__ bank,
    const float* __restrict__ bias, const float* __restrict__ critic_bias,
    float* __restrict__ out)
{
    __shared__ __bf16 As[BM][LDA];   // [m][k]
    __shared__ __bf16 Bs[BN][LDA];   // [n][k] (column-major B tile: contiguous K per column)

    const int tid  = threadIdx.x;
    const int m0   = blockIdx.x * BM;    // 6 M-tiles (fast grid dim -> L2 reuse of bank)
    const int n0   = blockIdx.y * BN;

    const int lane = tid & 31;
    const int wave = tid >> 5;
    const int wm   = (wave & 1) * 64;    // wave covers 64 rows x 32 cols
    const int wn   = (wave >> 1) * 32;
    const int l16  = lane & 15;
    const int koff = (lane < 16) ? 0 : 8;   // 16-bit A/B layout: lo lanes K{0-7,16-23}, hi K{8-15,24-31}
    const int mofs = (lane < 16) ? 0 : 8;   // C/D layout: hi lanes hold M+8

    AccF acc[4][2];
    #pragma unroll
    for (int i = 0; i < 4; ++i)
        #pragma unroll
        for (int j = 0; j < 2; ++j)
            #pragma unroll
            for (int v = 0; v < 8; ++v) acc[i][j].f[v] = 0.f;

    const int arow  = tid >> 1;          // 0..127
    const int ahalf = (tid & 1) * 16;    // element offset 0 / 16
    const int nr    = n0 + arow;
    const bool nrok = (nr < NDIM);
    const float* bsrc = bank + (size_t)nr * EDIM + ahalf;

    for (int k0 = 0; k0 < EDIM; k0 += BK) {
        // --- A tile: 128x32 bf16, 32B per thread ---
        {
            const uint4* src = (const uint4*)(Abig + (size_t)(m0 + arow) * EDIM + k0 + ahalf);
            uint4 v0 = src[0];
            uint4 v1 = src[1];
            *(uint4*)&As[arow][ahalf]     = v0;
            *(uint4*)&As[arow][ahalf + 8] = v1;
        }
        // --- B tile: 128 bank rows x 32 floats -> bf16, 64B per thread ---
        {
            float4 f0, f1, f2, f3;
            if (nrok) {
                const float4* src = (const float4*)(bsrc + k0);
                f0 = src[0]; f1 = src[1]; f2 = src[2]; f3 = src[3];
                if (k0 + 2 * BK < EDIM)                 // global_prefetch_b8 two steps ahead
                    __builtin_prefetch(bsrc + k0 + 2 * BK, 0, 1);
            } else {
                f0 = f1 = f2 = f3 = make_float4(0.f, 0.f, 0.f, 0.f);
            }
            uint4 p0, p1;
            p0.x = pack2bf(f0.x, f0.y); p0.y = pack2bf(f0.z, f0.w);
            p0.z = pack2bf(f1.x, f1.y); p0.w = pack2bf(f1.z, f1.w);
            p1.x = pack2bf(f2.x, f2.y); p1.y = pack2bf(f2.z, f2.w);
            p1.z = pack2bf(f3.x, f3.y); p1.w = pack2bf(f3.z, f3.w);
            *(uint4*)&Bs[arow][ahalf]     = p0;
            *(uint4*)&Bs[arow][ahalf + 8] = p1;
        }
        __syncthreads();

        FragBF afrag[4], bfrag[2];
        #pragma unroll
        for (int i = 0; i < 4; ++i) {
            const int m = wm + i * 16 + l16;
            afrag[i].u[0] = *(const uint4*)&As[m][koff];
            afrag[i].u[1] = *(const uint4*)&As[m][koff + 16];
        }
        #pragma unroll
        for (int j = 0; j < 2; ++j) {
            const int n = wn + j * 16 + l16;
            bfrag[j].u[0] = *(const uint4*)&Bs[n][koff];
            bfrag[j].u[1] = *(const uint4*)&Bs[n][koff + 16];
        }
        #pragma unroll
        for (int i = 0; i < 4; ++i)
            #pragma unroll
            for (int j = 0; j < 2; ++j)
                acc[i][j].v = __builtin_amdgcn_wmma_f32_16x16x32_bf16(
                    false, afrag[i].v, false, bfrag[j].v,
                    (short)0, acc[i][j].v, false, false);
        __syncthreads();
    }

    // --- store: D layout is VGPR v -> M = v (+8 for hi lanes), N = lane&15 ---
    const float b0 = bias[0];
    #pragma unroll
    for (int i = 0; i < 4; ++i) {
        const int head = (m0 + wm + i * 16) >> 8;             // 256 rows per head
        const float bv = (head == 0) ? b0 : critic_bias[head - 1];
        const int rbase = m0 + wm + i * 16 + mofs;
        #pragma unroll
        for (int j = 0; j < 2; ++j) {
            const int col = n0 + wn + j * 16 + l16;
            if (col < NDIM) {
                #pragma unroll
                for (int v = 0; v < 8; ++v) {
                    out[(size_t)(rbase + v) * NDIM + col] = acc[i][j].f[v] + bv;
                }
            }
        }
    }
}

// ---------------------------------------------------------------------------
// Kernel 4: mask previously used examples on the policy head:
//   out[0, b*T+t, idx[b,u]] = -inf  for u < t   (384 scattered writes)
// ---------------------------------------------------------------------------
__global__ void mask_kernel(const int* __restrict__ idx, float* __restrict__ out)
{
    const int id = blockIdx.x * blockDim.x + threadIdx.x;
    if (id >= BDIM * 6) return;
    const int b = id / 6, j = id % 6;
    const int tj[6] = {1, 2, 2, 3, 3, 3};
    const int uj[6] = {0, 0, 1, 0, 1, 2};
    const int t = tj[j], u = uj[j];
    const int col = idx[b * TDIM + u];
    out[(size_t)(b * TDIM + t) * NDIM + col] = -__builtin_inff();
}

// ---------------------------------------------------------------------------
extern "C" void kernel_launch(void* const* d_in, const int* in_sizes, int n_in,
                              void* d_out, int out_size, void* d_ws, size_t ws_size,
                              hipStream_t stream) {
    (void)in_sizes; (void)n_in; (void)out_size; (void)ws_size;
    const float* cur   = (const float*)d_in[0];
    const float* ex    = (const float*)d_in[1];
    const int*   pidx  = (const int*)  d_in[2];
    const float* bank  = (const float*)d_in[3];
    const float* w_in  = (const float*)d_in[4];
    const float* bil   = (const float*)d_in[5];
    const float* bias  = (const float*)d_in[6];
    const float* cbil  = (const float*)d_in[7];
    const float* cbias = (const float*)d_in[8];
    float* out = (float*)d_out;

    float*  lat  = (float*)d_ws;                                        // 64*4*768 f32
    __bf16* Abig = (__bf16*)((char*)d_ws + (size_t)BDIM*TDIM*HDIM*4);   // 768*768 bf16

    latent_kernel<<<dim3(HDIM / 256, BDIM), 256, 0, stream>>>(cur, ex, w_in, lat);
    aproj_kernel<<<MTOT, 256, 0, stream>>>(lat, bil, cbil, Abig);
    gemm_bank_kernel<<<dim3(MTOT / BM, (NDIM + BN - 1) / BN), 256, 0, stream>>>(
        Abig, bank, bias, cbias, out);
    mask_kernel<<<2, 256, 0, stream>>>(pidx, out);
}
